// CenterAwareSampling_21165598835477
// MI455X (gfx1250) — matrix operand order repaired and checked by reading, not compiled, not run on hardware
//
#include <hip/hip_runtime.h>
#include <hip/hip_fp16.h>

typedef __attribute__((ext_vector_type(16))) _Float16 v16h;
typedef __attribute__((ext_vector_type(8)))  float    v8f;

#define C_IN   128
#define HID    256
#define NPB    65536          // points per batch
#define NBATCH 8
#define TOTAL_ROWS (NBATCH * NPB)
#define TM_WG  128            // rows per workgroup (8 waves x 16)
#define SAMPLE 4096
#define XSTR   264            // padded f16 row stride for LDS activations

union V16H { v16h v; uint4 q[2]; };

// ---------------------------------------------------------------------------
// Prep: transpose weights to [N][K] f16, fold BN into scale/shift
// ---------------------------------------------------------------------------
__global__ __launch_bounds__(256) void prep_kernel(
    const float* __restrict__ W1, const float* __restrict__ W2,
    const float* __restrict__ g1, const float* __restrict__ b1,
    const float* __restrict__ m1, const float* __restrict__ v1,
    const float* __restrict__ g2, const float* __restrict__ b2,
    const float* __restrict__ m2, const float* __restrict__ v2,
    _Float16* __restrict__ W1t, _Float16* __restrict__ W2t,
    float* __restrict__ sc1, float* __restrict__ sh1,
    float* __restrict__ sc2, float* __restrict__ sh2)
{
    const int i = blockIdx.x * blockDim.x + threadIdx.x;
    if (i < HID * HID) {                 // W2 is (256,256) row-major [k][n]
        const int k = i >> 8, n = i & 255;
        W2t[n * HID + k] = (_Float16)W2[i];
    }
    if (i < C_IN * HID) {                // W1 is (128,256) row-major [k][n]
        const int k = i >> 8, n = i & 255;
        W1t[n * C_IN + k] = (_Float16)W1[i];
    }
    if (i < HID) {
        const float s1 = g1[i] * rsqrtf(v1[i] + 1e-5f);
        sc1[i] = s1; sh1[i] = b1[i] - m1[i] * s1;
        const float s2 = g2[i] * rsqrtf(v2[i] + 1e-5f);
        sc2[i] = s2; sh2[i] = b2[i] - m2[i] * s2;
    }
}

// ---------------------------------------------------------------------------
// Fused MLP: feats -> L1(WMMA f16, BN, ReLU) -> LDS -> L2(WMMA) -> head -> score
// ---------------------------------------------------------------------------
__global__ __launch_bounds__(256) void mlp_kernel(
    const float* __restrict__ feats,
    const _Float16* __restrict__ W1t,   // [256][128]
    const _Float16* __restrict__ W2t,   // [256][256]
    const float* __restrict__ sc1, const float* __restrict__ sh1,
    const float* __restrict__ sc2, const float* __restrict__ sh2,
    const float* __restrict__ W3, const float* __restrict__ b3,
    float* __restrict__ scores)
{
    extern __shared__ _Float16 lds[];           // 8 waves * 16 * XSTR f16
    const int tid  = threadIdx.x;
    const int wave = tid >> 5;
    const int lane = tid & 31;
    const int half = lane >> 4;                 // 0 or 1
    const int n0   = lane & 15;

    const int rowBase = blockIdx.x * TM_WG + wave * 16;
    const int mrow    = rowBase + n0;           // A-matrix row owned by this lane

    // ---- Layer 1 A fragments: global f32 -> f16, per documented 16x32 layout
    v16h A1[4];
    {
        const float* fr = feats + (size_t)mrow * C_IN;
        #pragma unroll
        for (int s = 0; s < 4; ++s) {
            const int kb0 = s * 32 + half * 8;
            const int kb1 = s * 32 + 16 + half * 8;
            float f[16];
            *(float4*)(f + 0)  = *(const float4*)(fr + kb0);
            *(float4*)(f + 4)  = *(const float4*)(fr + kb0 + 4);
            *(float4*)(f + 8)  = *(const float4*)(fr + kb1);
            *(float4*)(f + 12) = *(const float4*)(fr + kb1 + 4);
            v16h a;
            #pragma unroll
            for (int e = 0; e < 16; ++e) a[e] = (_Float16)f[e];
            A1[s] = a;
        }
    }

    _Float16* sX = lds + wave * (16 * XSTR);    // wave-private activation strip

    // ---- Layer 1: 16 column tiles of 16
    for (int t = 0; t < 16; ++t) {
        const int n = t * 16 + n0;
        v8f acc = {};
        #pragma unroll
        for (int s = 0; s < 4; ++s) {
            V16H bf;
            const _Float16* bp = W1t + n * C_IN + s * 32 + half * 8;
            bf.q[0] = *(const uint4*)(bp);
            bf.q[1] = *(const uint4*)(bp + 16);
            acc = __builtin_amdgcn_wmma_f32_16x16x32_f16(
                false, A1[s], false, bf.v, (short)0, acc, false, false);
        }
        const float sc = sc1[n], sh = sh1[n];
        #pragma unroll
        for (int r = 0; r < 8; ++r) {
            float v = acc[r] * sc + sh;
            v = v > 0.0f ? v : 0.0f;
            sX[(r + 8 * half) * XSTR + n] = (_Float16)v;
        }
    }
    __syncthreads();

    // ---- Layer 2 A fragments from own LDS strip (K = 256 -> 8 k-steps)
    v16h A2[8];
    {
        const _Float16* xr = sX + n0 * XSTR;    // row m = n0
        #pragma unroll
        for (int s = 0; s < 8; ++s) {
            V16H a;
            a.q[0] = *(const uint4*)(xr + s * 32 + half * 8);
            a.q[1] = *(const uint4*)(xr + s * 32 + 16 + half * 8);
            A2[s] = a.v;
        }
    }

    // ---- Layer 2 + fused head partials
    float p[3][8];
    #pragma unroll
    for (int c = 0; c < 3; ++c)
        #pragma unroll
        for (int r = 0; r < 8; ++r) p[c][r] = 0.0f;

    for (int t = 0; t < 16; ++t) {
        const int n = t * 16 + n0;
        v8f acc = {};
        #pragma unroll
        for (int s = 0; s < 8; ++s) {
            V16H bf;
            const _Float16* bp = W2t + n * HID + s * 32 + half * 8;
            bf.q[0] = *(const uint4*)(bp);
            bf.q[1] = *(const uint4*)(bp + 16);
            acc = __builtin_amdgcn_wmma_f32_16x16x32_f16(
                false, A2[s], false, bf.v, (short)0, acc, false, false);
        }
        const float sc  = sc2[n], sh = sh2[n];
        const float w30 = W3[n * 3 + 0];
        const float w31 = W3[n * 3 + 1];
        const float w32 = W3[n * 3 + 2];
        #pragma unroll
        for (int r = 0; r < 8; ++r) {
            float v = acc[r] * sc + sh;
            v = v > 0.0f ? v : 0.0f;
            p[0][r] += v * w30;
            p[1][r] += v * w31;
            p[2][r] += v * w32;
        }
    }

    // ---- reduce head partials across the 16 lanes sharing the same rows
    #pragma unroll
    for (int off = 8; off >= 1; off >>= 1)
        #pragma unroll
        for (int c = 0; c < 3; ++c)
            #pragma unroll
            for (int r = 0; r < 8; ++r)
                p[c][r] += __shfl_xor(p[c][r], off, 16);

    if (n0 == 0) {
        const float B0 = b3[0], B1 = b3[1], B2 = b3[2];
        #pragma unroll
        for (int r = 0; r < 8; ++r) {
            const float l0 = p[0][r] + B0;
            const float l1 = p[1][r] + B1;
            const float l2 = p[2][r] + B2;
            // sigmoid is monotone: rank by max-logit directly
            scores[rowBase + 8 * half + r] = fmaxf(fmaxf(l0, l1), l2);
        }
    }
}

// ---------------------------------------------------------------------------
// Deterministic exact top-4096 per batch: 64-bit tie-free keys, 8-pass radix
// select for the exact threshold, gather, bitonic sort descending, emit int64.
// ---------------------------------------------------------------------------
__global__ __launch_bounds__(1024) void select_kernel(
    const float* __restrict__ scores, long long* __restrict__ out)
{
    __shared__ unsigned int hist[256];
    __shared__ unsigned long long sPrefix;
    __shared__ unsigned int sQuota;
    __shared__ unsigned int cnt;
    __shared__ unsigned long long keys[SAMPLE];

    const int b   = blockIdx.x;
    const int tid = threadIdx.x;
    const float* s = scores + (size_t)b * NPB;

    auto makeKey = [&](int i) -> unsigned long long {
        unsigned int u = __float_as_uint(s[i]);
        u = (u & 0x80000000u) ? ~u : (u | 0x80000000u);   // monotone float map
        return ((unsigned long long)u << 32) | (unsigned int)(~(unsigned int)i);
    };

    if (tid == 0) { sPrefix = 0ull; sQuota = SAMPLE; }
    __syncthreads();

    for (int pass = 7; pass >= 0; --pass) {
        const int shift = pass * 8;
        for (int i = tid; i < 256; i += 1024) hist[i] = 0u;
        __syncthreads();
        const unsigned long long prefix = sPrefix;
        const unsigned int quota = sQuota;
        for (int i = tid; i < NPB; i += 1024) {
            const unsigned long long k = makeKey(i);
            if (pass == 7 || ((k >> (shift + 8)) == (prefix >> (shift + 8))))
                atomicAdd(&hist[(unsigned int)(k >> shift) & 255u], 1u);
        }
        __syncthreads();
        if (tid == 0) {
            unsigned int cum = 0;
            int d = 255;
            for (; d > 0; --d) {
                const unsigned int c = hist[d];
                if (cum + c >= quota) break;
                cum += c;
            }
            sPrefix = prefix | ((unsigned long long)(unsigned int)d << shift);
            sQuota  = quota - cum;
        }
        __syncthreads();
    }
    const unsigned long long thresh = sPrefix;  // exact 4096-th largest key

    if (tid == 0) cnt = 0u;
    __syncthreads();
    for (int i = tid; i < NPB; i += 1024) {
        const unsigned long long k = makeKey(i);
        if (k >= thresh) {
            const unsigned int pos = atomicAdd(&cnt, 1u);
            if (pos < SAMPLE) keys[pos] = k;
        }
    }
    __syncthreads();

    // bitonic sort, descending
    for (unsigned int kk = 2; kk <= SAMPLE; kk <<= 1) {
        for (unsigned int j = kk >> 1; j > 0; j >>= 1) {
            for (unsigned int ii = tid; ii < SAMPLE; ii += 1024) {
                const unsigned int partner = ii ^ j;
                if (partner > ii) {
                    const unsigned long long a = keys[ii];
                    const unsigned long long c = keys[partner];
                    const bool desc = ((ii & kk) == 0);
                    if (desc ? (a < c) : (a > c)) {
                        keys[ii] = c; keys[partner] = a;
                    }
                }
            }
            __syncthreads();
        }
    }

    for (int i = tid; i < SAMPLE; i += 1024)
        out[(size_t)b * SAMPLE + i] =
            (long long)(~(unsigned int)keys[i]);   // recover index from low32
}

// ---------------------------------------------------------------------------
extern "C" void kernel_launch(void* const* d_in, const int* in_sizes, int n_in,
                              void* d_out, int out_size, void* d_ws, size_t ws_size,
                              hipStream_t stream) {
    (void)in_sizes; (void)n_in; (void)out_size; (void)ws_size;
    const float* feats = (const float*)d_in[1];
    const float* W1 = (const float*)d_in[2];
    const float* g1 = (const float*)d_in[3];
    const float* b1 = (const float*)d_in[4];
    const float* m1 = (const float*)d_in[5];
    const float* v1 = (const float*)d_in[6];
    const float* W2 = (const float*)d_in[7];
    const float* g2 = (const float*)d_in[8];
    const float* b2 = (const float*)d_in[9];
    const float* m2 = (const float*)d_in[10];
    const float* v2 = (const float*)d_in[11];
    const float* W3 = (const float*)d_in[12];
    const float* b3 = (const float*)d_in[13];

    char* ws = (char*)d_ws;
    _Float16* W1t = (_Float16*)(ws);                 //  64 KB
    _Float16* W2t = (_Float16*)(ws + 65536);         // 128 KB
    float* sc1    = (float*)(ws + 196608);
    float* sh1    = (float*)(ws + 197632);
    float* sc2    = (float*)(ws + 198656);
    float* sh2    = (float*)(ws + 199680);
    float* scores = (float*)(ws + 200704);           // 2 MB

    prep_kernel<<<256, 256, 0, stream>>>(W1, W2, g1, b1, m1, v1,
                                         g2, b2, m2, v2,
                                         W1t, W2t, sc1, sh1, sc2, sh2);

    const size_t ldsBytes = 8u * 16u * XSTR * sizeof(_Float16);  // 67,584 B
    mlp_kernel<<<TOTAL_ROWS / TM_WG, 256, ldsBytes, stream>>>(
        feats, W1t, W2t, sc1, sh1, sc2, sh2, W3, b3, scores);

    select_kernel<<<NBATCH, 1024, 0, stream>>>(scores, (long long*)d_out);
}
